// AKT_50852412785099
// MI455X (gfx1250) — matrix operand order repaired and checked by reading, not compiled, not run on hardware
//
#include <hip/hip_runtime.h>
#include <hip/hip_bf16.h>

// Problem constants (fixed by reference)
#define B_ 4
#define S_ 512
#define P_ 5000
#define C_ 256
#define D_ 256
#define H_ 8
#define BS (B_ * S_)   // 2048 rows
#define HD (H_ * D_)   // 2048

// GEMM tiling
#define BM_ 128
#define BN_ 128
#define BK_ 64
#define LDSW (BK_ + 8)   // 72 halves = 144B row pitch (TDM pad: 16B per 128B row)

typedef __attribute__((ext_vector_type(16))) _Float16 v16h;
typedef __attribute__((ext_vector_type(8)))  _Float16 v8h;
typedef __attribute__((ext_vector_type(8)))  float    v8f;
typedef unsigned int u32x4 __attribute__((ext_vector_type(4)));
typedef int          i32x8 __attribute__((ext_vector_type(8)));
typedef int          i32x4 __attribute__((ext_vector_type(4)));

#if __has_builtin(__builtin_amdgcn_tensor_load_to_lds)
#define USE_TDM 1
#else
#define USE_TDM 0
#endif

#if USE_TDM
// Build a D# (ISA ch.8) for a 2D 128x64-halves tile and fire the TDM.
// Group0: count=1 | lds_addr | global_addr | type=2
// Group1: data_size=2B, pad_enable (32 DWORD interval, 4 DWORD pad -> 144B pitch),
//         tile_dim0=64, tile_dim1=128, dim0_stride=ld (elements)
__device__ __forceinline__ void tdm_tile_load(const _Float16* g, int ld, unsigned ldsOff)
{
    unsigned long long ga = (unsigned long long)(size_t)g;
    u32x4 g0;
    g0[0] = 1u;                                                 // count = 1
    g0[1] = ldsOff;                                             // lds_addr (bytes)
    g0[2] = (unsigned)ga;                                       // global_addr[31:0]
    g0[3] = (unsigned)((ga >> 32) & 0x01FFFFFFu) | (2u << 30);  // [56:32] | type=2
    i32x8 g1;
    g1[0] = (int)((1u << 16) |   // data_size = 2 bytes
                  (1u << 20) |   // pad_enable
                  (4u << 22) |   // pad_interval code 4 = 32 DWORDs (128B)
                  (3u << 25));   // pad_amount  code 3 = 4 DWORDs  (16B)
    unsigned td0 = 1u << 20, td1 = 1u << 20;   // huge tensor dims: never OOB-clip
    g1[1] = (int)((td0 & 0xFFFFu) << 16);                        // [63:48]=td0 lo16
    g1[2] = (int)((td0 >> 16) | ((td1 & 0xFFFFu) << 16));        // td0 hi16 | td1 lo16
    g1[3] = (int)((td1 >> 16) | ((unsigned)BK_ << 16));          // td1 hi16 | tile_dim0
    g1[4] = (int)BM_;                                            // tile_dim1 | tile_dim2=0
    unsigned long long st = (unsigned long long)ld;              // dim0 stride (elems)
    g1[5] = (int)(st & 0xFFFFFFFFu);
    g1[6] = (int)((st >> 32) & 0xFFFFu);                         // dim1_stride = 0
    g1[7] = 0;
    i32x4 z4 = {0, 0, 0, 0};
#if __has_include(<hip/amd_detail/amd_gfx1250_TDM.h>)
    // amdgpu-toolchain (clang-23 / therock-10.0): 6-arg builtin
    i32x8 z8 = {0, 0, 0, 0, 0, 0, 0, 0};
    __builtin_amdgcn_tensor_load_to_lds(g0, g1, z4, z4, z8, 0);
#else
    // ROCm 7.2 (clang-22): 5-arg builtin
    __builtin_amdgcn_tensor_load_to_lds(g0, g1, z4, z4, 0);
#endif
}
#endif

// ---------------------------------------------------------------------------
// Batched WMMA GEMM: C[M,N] = alpha * A[M,K] @ B[K,N] (+ C if accum)
// B supplied TRANSPOSED (Bt[n,k]). Block tile 128x128, K-chunk 64, TDM
// double-buffered LDS staging. 8 waves as 4(M)x2(N); each wave 32x64 =
// 2x4 accumulators = 8 WMMAs per 32-K step.
// Batch offsets: off = (z/div)*outer + (z%div)*inner.
// ---------------------------------------------------------------------------
__global__ __launch_bounds__(256) void wmma_gemm_kernel(
    const _Float16* __restrict__ A, const _Float16* __restrict__ Bt,
    float* __restrict__ Cf, _Float16* __restrict__ Ch,
    int M, int N, int K, int lda, int ldb, int ldc,
    int divA, long long outA, long long inA,
    int divB, long long outB, long long inB,
    int divC, long long outC, long long inC,
    float alpha, int accum, int store_f16)
{
    __shared__ _Float16 smA[2][BM_ * LDSW];
    __shared__ _Float16 smB[2][BN_ * LDSW];

    int z = blockIdx.z;
    A  += (long long)(z / divA) * outA + (long long)(z % divA) * inA;
    Bt += (long long)(z / divB) * outB + (long long)(z % divB) * inB;
    long long coff = (long long)(z / divC) * outC + (long long)(z % divC) * inC;

    int lane = threadIdx.x & 31, wave = threadIdx.x >> 5;
    int r  = lane & 15;          // row-in-frag (A) / col-in-frag (B)
    int kh = lane >> 4;          // half-wave K selector
    int mw = (wave & 3) * 32;    // wave M offset within block tile
    int nw = (wave >> 2) * 64;   // wave N offset within block tile
    long long mBlk = (long long)blockIdx.y * BM_;
    long long nBlk = (long long)blockIdx.x * BN_;

    v8f acc[2][4] = {};
    int nch = K / BK_;

#if USE_TDM
    if (threadIdx.x < 32) {   // one TDM issue per op (EXEC ignored; wave0 only)
        tdm_tile_load(A  + mBlk * lda, lda, (unsigned)(size_t)&smA[0][0]);
        tdm_tile_load(Bt + nBlk * ldb, ldb, (unsigned)(size_t)&smB[0][0]);
    }
#endif
    int buf = 0;
    for (int c = 0; c < nch; ++c) {
#if USE_TDM
        if (threadIdx.x < 32) {
            if (c + 1 < nch) {
                long long k2 = (long long)(c + 1) * BK_;
                tdm_tile_load(A  + mBlk * lda + k2, lda, (unsigned)(size_t)&smA[buf ^ 1][0]);
                tdm_tile_load(Bt + nBlk * ldb + k2, ldb, (unsigned)(size_t)&smB[buf ^ 1][0]);
                __builtin_amdgcn_s_wait_tensorcnt(2);  // current chunk's 2 ops done
            } else {
                __builtin_amdgcn_s_wait_tensorcnt(0);
            }
        }
        __syncthreads();
#else
        {   // portable fallback: cooperative synchronous staging
            long long k0 = (long long)c * BK_;
            int t = threadIdx.x;
#pragma unroll
            for (int i = 0; i < 4; ++i) {          // 1024 v8h chunks per tile
                int idx = t + i * 256;
                int row = idx >> 3, c8 = (idx & 7) * 8;
                *(v8h*)&smA[buf][row * LDSW + c8] =
                    *(const v8h*)(A + (mBlk + row) * lda + k0 + c8);
                *(v8h*)&smB[buf][row * LDSW + c8] =
                    *(const v8h*)(Bt + (nBlk + row) * ldb + k0 + c8);
            }
            __syncthreads();
        }
#endif
        const _Float16* As = &smA[buf][mw * LDSW];
        const _Float16* Bs = &smB[buf][nw * LDSW];
#pragma unroll
        for (int ks = 0; ks < BK_; ks += 32) {
            v16h af[2];
#pragma unroll
            for (int mf = 0; mf < 2; ++mf) {
                const _Float16* ap = As + (mf * 16 + r) * LDSW + ks;
                v8h lo = *(const v8h*)(ap + kh * 8);
                v8h hi = *(const v8h*)(ap + 16 + kh * 8);
#pragma unroll
                for (int e = 0; e < 8; ++e) { af[mf][e] = lo[e]; af[mf][e + 8] = hi[e]; }
            }
#pragma unroll
            for (int nf = 0; nf < 4; ++nf) {
                const _Float16* bp = Bs + (nf * 16 + r) * LDSW + ks + kh * 8;
                v8h blo = *(const v8h*)bp;
                v8h bhi = *(const v8h*)(bp + 16);
                v16h bf;
#pragma unroll
                for (int e = 0; e < 8; ++e) { bf[e] = blo[e]; bf[e + 8] = bhi[e]; }
#pragma unroll
                for (int mf = 0; mf < 2; ++mf)
                    acc[mf][nf] = __builtin_amdgcn_wmma_f32_16x16x32_f16(
                        false, af[mf], false, bf, (short)0, acc[mf][nf], false, false);
            }
        }
        __syncthreads();
        buf ^= 1;
    }

    // store: C/D 16x16 layout -> lane(0-15)=N, vgpr rr = M=rr (+8 for hi lanes)
#pragma unroll
    for (int mf = 0; mf < 2; ++mf) {
#pragma unroll
        for (int nf = 0; nf < 4; ++nf) {
#pragma unroll
            for (int rr = 0; rr < 8; ++rr) {
                long long m = mBlk + mw + mf * 16 + kh * 8 + rr;
                long long n = nBlk + nw + nf * 16 + r;
                size_t idx = (size_t)(coff + m * ldc + n);
                float v = alpha * acc[mf][nf][rr];
                if (store_f16) {
                    Ch[idx] = (_Float16)v;
                } else {
                    if (accum) v += Cf[idx];
                    Cf[idx] = v;
                }
            }
        }
    }
}

// ---------------------------------------------------------------------------
// Block-wide reductions (256 threads)
// ---------------------------------------------------------------------------
__device__ __forceinline__ float blockMax(float v, float* red) {
    int t = threadIdx.x;
    red[t] = v; __syncthreads();
    for (int s = 128; s > 0; s >>= 1) {
        if (t < s) red[t] = fmaxf(red[t], red[t + s]);
        __syncthreads();
    }
    float r = red[0]; __syncthreads();
    return r;
}
__device__ __forceinline__ float blockSum(float v, float* red) {
    int t = threadIdx.x;
    red[t] = v; __syncthreads();
    for (int s = 128; s > 0; s >>= 1) {
        if (t < s) red[t] += red[t + s];
        __syncthreads();
    }
    float r = red[0]; __syncthreads();
    return r;
}

// ---------------------------------------------------------------------------
// Fused temporal-decay attention weights. One block per (b,h,i) row.
// softmax -> tril -> renorm -> cumsum -> decay -> rescore -> softmax -> renorm
// ---------------------------------------------------------------------------
__global__ __launch_bounds__(256) void attn_row_kernel(
    const float* __restrict__ scores, const float* __restrict__ tsf,
    const float* __restrict__ theta, _Float16* __restrict__ aOut, int strict)
{
    __shared__ float red[256];
    int i = blockIdx.x, h = blockIdx.y, b = blockIdx.z;
    int t = threadIdx.x;
    size_t base = (((size_t)(b * H_ + h)) * S_ + i) * S_;
    int j0 = 2 * t, j1 = 2 * t + 1;

    float sc0 = scores[base + j0];
    float sc1 = scores[base + j1];
    float m0 = (strict ? (j0 < i) : (j0 <= i)) ? 1.f : 0.f;
    float m1 = (strict ? (j1 < i) : (j1 <= i)) ? 1.f : 0.f;

    float mx = blockMax(fmaxf(sc0, sc1), red);
    float e0 = __expf(sc0 - mx), e1 = __expf(sc1 - mx);
    float sm = blockSum(e0 + e1, red);
    float x0 = (e0 / sm) * m0, x1 = (e1 / sm) * m1;
    float xs = blockSum(x0 + x1, red);
    float inv = (xs == 0.f) ? 0.f : 1.f / xs;
    x0 *= inv; x1 *= inv;

    // inclusive cumsum over 512 elems (pairwise + Hillis-Steele over 256)
    float pair = x0 + x1;
    red[t] = pair; __syncthreads();
    for (int off = 1; off < 256; off <<= 1) {
        float add = (t >= off) ? red[t - off] : 0.f;
        __syncthreads();
        red[t] += add;
        __syncthreads();
    }
    float excl = red[t] - pair;
    __syncthreads();
    float cum0 = excl + x0;
    float cum1 = cum0 + x1;

    float ti  = tsf[b * S_ + i];
    float th  = theta[0];
    float th2 = th * th;
    float d0 = (1.f - cum0) * (ti - tsf[b * S_ + j0]) * m0;
    float d1 = (1.f - cum1) * (ti - tsf[b * S_ + j1]) * m1;
    float s0 = sc0 * __expf(-th2 * d0);
    float s1 = sc1 * __expf(-th2 * d1);

    float mx2 = blockMax(fmaxf(s0, s1), red);
    float f0 = __expf(s0 - mx2), f1 = __expf(s1 - mx2);
    float sm2 = blockSum(f0 + f1, red);
    float a0 = (f0 / sm2) * m0, a1 = (f1 / sm2) * m1;
    float as = blockSum(a0 + a1, red);
    float inv2 = (as == 0.f) ? 0.f : 1.f / as;
    aOut[base + j0] = (_Float16)(a0 * inv2);
    aOut[base + j1] = (_Float16)(a1 * inv2);
}

// ---------------------------------------------------------------------------
// Embedding / small kernels
// ---------------------------------------------------------------------------
__global__ __launch_bounds__(256) void concept_kernel(
    const int* __restrict__ item, const float* __restrict__ Qm,
    const float* __restrict__ muq,
    _Float16* __restrict__ conceptH, _Float16* __restrict__ cmuH,
    float* __restrict__ cn)
{
    __shared__ float red[256];
    int row = blockIdx.x, c = threadIdx.x;
    int it = item[row] - 1;
    float cv = (it >= 0) ? Qm[(size_t)it * C_ + c] : 0.f;
    conceptH[(size_t)row * C_ + c] = (_Float16)cv;
    cmuH[(size_t)row * C_ + c]     = (_Float16)(cv * muq[c]);
    red[c] = cv; __syncthreads();
    for (int s = 128; s > 0; s >>= 1) { if (c < s) red[c] += red[c + s]; __syncthreads(); }
    if (c == 0) cn[row] = red[0];
}

__global__ void y_init_kernel(const int* __restrict__ correct,
                              const float* __restrict__ rEmb,
                              const float* __restrict__ cn,
                              float* __restrict__ yF)
{
    int row = blockIdx.x, d = threadIdx.x;
    yF[(size_t)row * D_ + d] = rEmb[(size_t)correct[row] * D_ + d] * cn[row];
}

__global__ void i32_to_f32_kernel(const int* in, float* out, int n) {
    int i = blockIdx.x * 256 + threadIdx.x;
    if (i < n) out[i] = (float)in[i];
}
__global__ void f32_to_f16_kernel(const float* in, _Float16* out, int n) {
    int i = blockIdx.x * 256 + threadIdx.x;
    if (i < n) out[i] = (_Float16)in[i];
}
// batched via blockIdx.z: out[z][c][r] = in[z][r][c]
__global__ void transpose_f32_f16_kernel(const float* in, _Float16* out,
                                         int rows, int cols) {
    int idx = blockIdx.x * 256 + threadIdx.x;
    int nz = rows * cols;
    if (idx < nz) {
        long long base = (long long)blockIdx.z * nz;
        int r = idx / cols, c = idx % cols;
        out[base + (long long)c * rows + r] = (_Float16)in[base + idx];
    }
}
__global__ void transpose_f16_kernel(const _Float16* in, _Float16* out,
                                     int rows, int cols) {
    int idx = blockIdx.x * 256 + threadIdx.x;
    int nz = rows * cols;
    if (idx < nz) {
        long long base = (long long)blockIdx.z * nz;
        int r = idx / cols, c = idx % cols;
        out[base + (long long)c * rows + r] = in[base + idx];
    }
}
// ret [B,H,S,D] f16 -> retC [B,S,H*D] f16
__global__ void permute_ret_kernel(const _Float16* in, _Float16* out) {
    long long idx = (long long)blockIdx.x * 256 + threadIdx.x;
    int d = (int)(idx % D_); long long t1 = idx / D_;
    int s = (int)(t1 % S_);  long long t2 = t1 / S_;
    int h = (int)(t2 % H_);  int b = (int)(t2 / H_);
    out[(((long long)b * S_ + s) * H_ + h) * D_ + d] = in[idx];
}

// pred = sigmoid(concat(out(row0 zeroed), x_hat) @ Wd + bd)
__global__ __launch_bounds__(256) void pred_kernel(
    const float* __restrict__ outF, const float* __restrict__ xhatF,
    const float* __restrict__ Wd, const float* __restrict__ bd,
    float* __restrict__ pred)
{
    __shared__ float red[256];
    int row = blockIdx.x, t = threadIdx.x;
    int s = row & (S_ - 1);
    float o = (s == 0) ? 0.f : outF[(size_t)row * D_ + t];
    float acc = o * Wd[t] + xhatF[(size_t)row * D_ + t] * Wd[D_ + t];
    red[t] = acc; __syncthreads();
    for (int st = 128; st > 0; st >>= 1) { if (t < st) red[t] += red[t + st]; __syncthreads(); }
    if (t == 0) pred[row] = 1.f / (1.f + __expf(-(red[0] + bd[0])));
}

// ---------------------------------------------------------------------------
// Host-side helpers
// ---------------------------------------------------------------------------
static inline void gemm(hipStream_t st, const _Float16* A, const _Float16* Bt,
                        float* Cf, _Float16* Ch,
                        int M, int N, int K, int lda, int ldb, int ldc, int batch,
                        int divA, long long oA, long long iA,
                        int divB, long long oB, long long iB,
                        int divC, long long oC, long long iC,
                        float alpha, int accum, int f16out)
{
    dim3 g(N / BN_, M / BM_, batch), b(256);
    wmma_gemm_kernel<<<g, b, 0, st>>>(A, Bt, Cf, Ch, M, N, K, lda, ldb, ldc,
                                      divA, oA, iA, divB, oB, iB, divC, oC, iC,
                                      alpha, accum, f16out);
}

static void run_mha(hipStream_t st, const float* qxF, const float* vxF,
                    const _Float16* wQt, const _Float16* wKt,
                    const _Float16* wVt, const _Float16* wOt,
                    const float* theta, const float* tsf, int strict, float* outF,
                    _Float16* qxh, _Float16* vxh, _Float16* Qh, _Float16* Kh,
                    _Float16* Vh, _Float16* Vt, float* scoresF, _Float16* aH,
                    _Float16* retH, _Float16* retC)
{
    const long long SD = (long long)S_ * D_;
    const long long DD = (long long)D_ * D_;
    const long long SS = (long long)S_ * S_;
    f32_to_f16_kernel<<<(BS * D_ + 255) / 256, 256, 0, st>>>(qxF, qxh, BS * D_);
    f32_to_f16_kernel<<<(BS * D_ + 255) / 256, 256, 0, st>>>(vxF, vxh, BS * D_);
    // Q/K/V projections, batched over z = b*H + h; f16 output
    gemm(st, qxh, wQt, nullptr, Qh, S_, D_, D_, D_, D_, D_, B_ * H_,
         H_, SD, 0, H_, 0, DD, 1, SD, 0, 1.f, 0, 1);
    gemm(st, qxh, wKt, nullptr, Kh, S_, D_, D_, D_, D_, D_, B_ * H_,
         H_, SD, 0, H_, 0, DD, 1, SD, 0, 1.f, 0, 1);
    gemm(st, vxh, wVt, nullptr, Vh, S_, D_, D_, D_, D_, D_, B_ * H_,
         H_, SD, 0, H_, 0, DD, 1, SD, 0, 1.f, 0, 1);
    // V -> V^T [D,S] per (b,h): AV GEMM's Bt operand becomes contiguous
    transpose_f16_kernel<<<dim3((S_ * D_ + 255) / 256, 1, B_ * H_), 256, 0, st>>>(
        Vh, Vt, S_, D_);
    // scores = Q @ K^T / sqrt(D)  (Bt == K natural layout)
    gemm(st, Qh, Kh, scoresF, nullptr, S_, S_, D_, D_, D_, S_, B_ * H_,
         1, SD, 0, 1, SD, 0, 1, SS, 0, 0.0625f, 0, 0);
    attn_row_kernel<<<dim3(S_, H_, B_), 256, 0, st>>>(scoresF, tsf, theta, aH, strict);
    // ret = a @ V  (Bt == V^T), f16 output
    gemm(st, aH, Vt, nullptr, retH, S_, D_, S_, S_, S_, D_, B_ * H_,
         1, SS, 0, 1, SD, 0, 1, SD, 0, 1.f, 0, 1);
    permute_ret_kernel<<<(B_ * H_ * S_ * D_ + 255) / 256, 256, 0, st>>>(retH, retC);
    // out = retC @ wO
    gemm(st, retC, wOt, outF, nullptr, BS, D_, HD, HD, HD, D_, 1,
         1, 0, 0, 1, 0, 0, 1, 0, 0, 1.f, 0, 0);
}

extern "C" void kernel_launch(void* const* d_in, const int* in_sizes, int n_in,
                              void* d_out, int out_size, void* d_ws, size_t ws_size,
                              hipStream_t stream)
{
    (void)in_sizes; (void)n_in; (void)out_size; (void)ws_size;
    const int*   item = (const int*)d_in[0];
    const int*   tsI  = (const int*)d_in[1];
    const int*   corr = (const int*)d_in[2];
    const float* Qm   = (const float*)d_in[3];
    const float* cEmb = (const float*)d_in[4];
    const float* dEmb = (const float*)d_in[5];
    const float* fEmb = (const float*)d_in[6];
    const float* muq  = (const float*)d_in[7];
    const float* rEmb = (const float*)d_in[8];
    const float* Wd   = (const float*)d_in[9];
    const float* bd   = (const float*)d_in[10];
    const float* wQ[3] = {(const float*)d_in[11], (const float*)d_in[16], (const float*)d_in[21]};
    const float* wK[3] = {(const float*)d_in[12], (const float*)d_in[17], (const float*)d_in[22]};
    const float* wV[3] = {(const float*)d_in[13], (const float*)d_in[18], (const float*)d_in[23]};
    const float* wO[3] = {(const float*)d_in[14], (const float*)d_in[19], (const float*)d_in[24]};
    const float* th[3] = {(const float*)d_in[15], (const float*)d_in[20], (const float*)d_in[25]};

    // bump allocator over d_ws (working set ~120 MB; fits the 192 MB L2)
    char* wsp = (char*)d_ws; size_t off = 0;
    auto alloc = [&](size_t bytes) -> void* {
        void* p = (void*)(wsp + off);
        off += (bytes + 255) & ~(size_t)255;
        return p;
    };
    _Float16* conceptH = (_Float16*)alloc((size_t)BS * C_ * 2);
    _Float16* cmuH     = (_Float16*)alloc((size_t)BS * C_ * 2);
    float* cn    = (float*)alloc((size_t)BS * 4);
    float* tsf   = (float*)alloc((size_t)BS * 4);
    float* xF    = (float*)alloc((size_t)BS * D_ * 4);
    float* yF    = (float*)alloc((size_t)BS * D_ * 4);
    float* xhatF = (float*)alloc((size_t)BS * D_ * 4);
    float* yhatF = (float*)alloc((size_t)BS * D_ * 4);
    float* outF  = (float*)alloc((size_t)BS * D_ * 4);
    _Float16* cET = (_Float16*)alloc((size_t)C_ * D_ * 2);
    _Float16* dET = (_Float16*)alloc((size_t)C_ * D_ * 2);
    _Float16* fET = (_Float16*)alloc((size_t)C_ * D_ * 2);
    _Float16 *wQt[3], *wKt[3], *wVt[3], *wOt[3];
    for (int p = 0; p < 3; ++p) {
        wQt[p] = (_Float16*)alloc((size_t)H_ * D_ * D_ * 2);
        wKt[p] = (_Float16*)alloc((size_t)H_ * D_ * D_ * 2);
        wVt[p] = (_Float16*)alloc((size_t)H_ * D_ * D_ * 2);
        wOt[p] = (_Float16*)alloc((size_t)HD * D_ * 2);
    }
    // shared MHA scratch (reused across the three attention blocks)
    _Float16* qxh = (_Float16*)alloc((size_t)BS * D_ * 2);
    _Float16* vxh = (_Float16*)alloc((size_t)BS * D_ * 2);
    _Float16* Qh  = (_Float16*)alloc((size_t)B_ * H_ * S_ * D_ * 2);
    _Float16* Kh  = (_Float16*)alloc((size_t)B_ * H_ * S_ * D_ * 2);
    _Float16* Vh  = (_Float16*)alloc((size_t)B_ * H_ * S_ * D_ * 2);
    _Float16* Vt  = (_Float16*)alloc((size_t)B_ * H_ * S_ * D_ * 2);
    float*    scoresF = (float*)alloc((size_t)B_ * H_ * S_ * S_ * 4);
    _Float16* aH   = (_Float16*)alloc((size_t)B_ * H_ * S_ * S_ * 2);
    _Float16* retH = (_Float16*)alloc((size_t)B_ * H_ * S_ * D_ * 2);
    _Float16* retC = (_Float16*)alloc((size_t)BS * HD * 2);

    // --- embeddings / precision conversion ---
    i32_to_f32_kernel<<<(BS + 255) / 256, 256, 0, stream>>>(tsI, tsf, BS);
    concept_kernel<<<BS, 256, 0, stream>>>(item, Qm, muq, conceptH, cmuH, cn);
    transpose_f32_f16_kernel<<<dim3((C_ * D_ + 255) / 256, 1, 1), 256, 0, stream>>>(cEmb, cET, C_, D_);
    transpose_f32_f16_kernel<<<dim3((C_ * D_ + 255) / 256, 1, 1), 256, 0, stream>>>(dEmb, dET, C_, D_);
    transpose_f32_f16_kernel<<<dim3((C_ * D_ + 255) / 256, 1, 1), 256, 0, stream>>>(fEmb, fET, C_, D_);
    for (int p = 0; p < 3; ++p) {
        transpose_f32_f16_kernel<<<dim3((D_ * D_ + 255) / 256, 1, H_), 256, 0, stream>>>(wQ[p], wQt[p], D_, D_);
        transpose_f32_f16_kernel<<<dim3((D_ * D_ + 255) / 256, 1, H_), 256, 0, stream>>>(wK[p], wKt[p], D_, D_);
        transpose_f32_f16_kernel<<<dim3((D_ * D_ + 255) / 256, 1, H_), 256, 0, stream>>>(wV[p], wVt[p], D_, D_);
        transpose_f32_f16_kernel<<<dim3((HD * D_ + 255) / 256, 1, 1), 256, 0, stream>>>(wO[p], wOt[p], HD, D_);
    }

    // x = concept @ c_embed + cmu @ d_embed
    gemm(stream, conceptH, cET, xF, nullptr, BS, D_, C_, C_, C_, D_, 1,
         1, 0, 0, 1, 0, 0, 1, 0, 0, 1.f, 0, 0);
    gemm(stream, cmuH, dET, xF, nullptr, BS, D_, C_, C_, C_, D_, 1,
         1, 0, 0, 1, 0, 0, 1, 0, 0, 1.f, 1, 0);
    // y = r_embed[correct]*cn + cmu @ f_embed
    y_init_kernel<<<BS, 256, 0, stream>>>(corr, rEmb, cn, yF);
    gemm(stream, cmuH, fET, yF, nullptr, BS, D_, C_, C_, C_, D_, 1,
         1, 0, 0, 1, 0, 0, 1, 0, 0, 1.f, 1, 0);

    // three temporal-decay MHA blocks
    run_mha(stream, xF, xF, wQt[0], wKt[0], wVt[0], wOt[0], th[0], tsf, 0, xhatF,
            qxh, vxh, Qh, Kh, Vh, Vt, scoresF, aH, retH, retC);
    run_mha(stream, yF, yF, wQt[1], wKt[1], wVt[1], wOt[1], th[1], tsf, 0, yhatF,
            qxh, vxh, Qh, Kh, Vh, Vt, scoresF, aH, retH, retC);
    run_mha(stream, xhatF, yhatF, wQt[2], wKt[2], wVt[2], wOt[2], th[2], tsf, 1, outF,
            qxh, vxh, Qh, Kh, Vh, Vt, scoresF, aH, retH, retC);

    // pred = sigmoid(concat(out(row0=0), x_hat) @ Wd + bd)
    pred_kernel<<<BS, 256, 0, stream>>>(outF, xhatF, Wd, bd, (float*)d_out);
}
